// lstmm_53377853555284
// MI455X (gfx1250) — matrix operand order repaired and checked by reading, not compiled, not run on hardware
//
#include <hip/hip_runtime.h>
#include <hip/hip_bf16.h>

// ---------------------------------------------------------------------------
// LSTM (B=256, T=2048, D=72, H=256) + linear heads, CDNA5 / gfx1250.
// Persistent-kernel recurrence on v_wmma_f32_16x16x32_bf16.
//   * fused input GEMM (no gx materialization)
//   * weights staged once into LDS (bank-conflict-free padded layout)
//   * anti-hoist opaque offset keeps fragment loads inside the t-loop
//   * depth-1 software pipeline on B fragments (hide ds latency)
//   * x(t+1) loaded/converted into registers BEFORE the grid barrier
//   * double-buffered bf16 h, c held in registers for all 2048 steps
// ---------------------------------------------------------------------------

#define HID  256
#define BAT  256
#define TSEQ 2048
#define DIN  72
#define DPAD 96
#define G4   1024       // 4*H
#define NBLK 64         // 4 B-tiles (64 rows) x 16 H-col tiles (16 cols)
#define NTHR 128        // 4 waves; wave r owns rows 16r, all 4 gates

#define WHH_LD 264      // 256 + 8 pad  -> 132 dwords/row, conflict-free
#define WIH_LD 104      // 96 + 8 pad   -> 52 dwords/row, conflict-free

typedef __attribute__((ext_vector_type(16))) __bf16 v16bf;
typedef __attribute__((ext_vector_type(8)))  __bf16 v8bf;
typedef __attribute__((ext_vector_type(8)))  float  v8f;

union FragAB { v16bf v; v8bf h[2]; };
union BF2    { unsigned u; __bf16 b[2]; };

// A-matrix 16x32 bf16 fragment (ISA 7.12.2): lane<16 -> row m=lane, K=0..7 &
// 16..23; lanes 16..31 -> same rows, K=8..15 & 24..31.
__device__ __forceinline__ v16bf load_frag_a(const __bf16* base, int ld, int lane) {
  const int m  = lane & 15;
  const int kb = (lane & 16) ? 8 : 0;
  const __bf16* p = base + m * ld + kb;
  FragAB f;
  f.h[0] = *(const v8bf*)(p);
  f.h[1] = *(const v8bf*)(p + 16);
  return f.v;
}

// B-matrix 32x16 bf16 fragment: lane -> column n=lane&15; lanes<16 cover
// K=0..7 & 16..23, lanes>=16 cover K=8..15 & 24..31 (matches A fragment).
__device__ __forceinline__ v16bf load_frag_b(const __bf16* base, int ld, int lane) {
  const int n  = lane & 15;
  const int kb = (lane & 16) ? 8 : 0;
  const __bf16* p = base + n * ld + kb;
  FragAB f;
  f.h[0] = *(const v8bf*)(p);
  f.h[1] = *(const v8bf*)(p + 16);
  return f.v;
}

__device__ __forceinline__ float fsig(float x)  { return 1.0f / (1.0f + __expf(-x)); }
__device__ __forceinline__ float ftanh(float x) { return 2.0f / (1.0f + __expf(-2.0f * x)) - 1.0f; }

// ---------------------------------------------------------------------------
// Prep: fp32 -> bf16 weights (W_ih zero-padded K 72->96), bias fold, h0 -> buf0.
// ---------------------------------------------------------------------------
__global__ void lstm_prep_kernel(const float* __restrict__ wih,
                                 const float* __restrict__ whh,
                                 const float* __restrict__ b_ih,
                                 const float* __restrict__ b_hh,
                                 const float* __restrict__ h0,
                                 __bf16* __restrict__ wih_b,
                                 __bf16* __restrict__ whh_b,
                                 float*  __restrict__ bias,
                                 __bf16* __restrict__ hbuf0) {
  const int i = blockIdx.x * blockDim.x + threadIdx.x;
  if (i < G4 * HID)  whh_b[i] = (__bf16)whh[i];
  if (i < G4 * DPAD) {
    const int row = i / DPAD, k = i % DPAD;
    wih_b[i] = (k < DIN) ? (__bf16)wih[row * DIN + k] : (__bf16)0.0f;
  }
  if (i < BAT * HID) hbuf0[i] = (__bf16)h0[i];
  if (i < G4)        bias[i]  = b_ih[i] + b_hh[i];
}

// ---------------------------------------------------------------------------
// Persistent LSTM recurrence. Block bx: batch rows b0..b0+63, h cols hc0..hc0+15.
// Wave r (0..3): rows b0+16r, all 4 gate tiles -> i,f,g,o register-local.
// ---------------------------------------------------------------------------
__global__ __launch_bounds__(NTHR, 1)
void lstm_persistent_kernel(const float* __restrict__ x,     // [B][T][D] fp32
                            const float* __restrict__ c0,    // [B][H]
                            const __bf16* __restrict__ wih,  // [4H][96] bf16
                            const __bf16* __restrict__ whh,  // [4H][H]  bf16
                            const float*  __restrict__ bias, // [4H]
                            __bf16* __restrict__ hbuf,       // [2][B][H] bf16
                            float*  __restrict__ hnf,        // [B][H]
                            float*  __restrict__ cnf,        // [B][H]
                            unsigned* __restrict__ sync_cnt) {
  // 33792 + 13312 + 13312 = 60416 bytes of LDS
  __shared__ __align__(16) __bf16 whh_s[64][WHH_LD];  // local gate col = s*16+c
  __shared__ __align__(16) __bf16 wih_s[64][WIH_LD];
  __shared__ __align__(16) __bf16 xs[64][WIH_LD];     // x_t tile, K padded to 96

  const int bx   = blockIdx.x;
  const int b0   = (bx >> 4) * 64;    // batch-row tile
  const int hc0  = (bx & 15) * 16;    // h-column tile
  const int tid  = threadIdx.x;
  const int wv   = tid >> 5;          // wave = row sub-tile 0..3
  const int lane = tid & 31;
  const int nlo  = lane & 15;
  const int mhi  = (lane & 16) ? 8 : 0;

  // ---- one-time LDS fill: weight slices (64 gate-cols owned by this block) --
  {
    const int row  = tid >> 1;        // 0..63 local gate-col index
    const int half = tid & 1;
    const int s = row >> 4, c = row & 15;
    const int grow = s * HID + hc0 + c;
    const __bf16* gh = whh + (size_t)grow * HID + half * 128;
    __bf16* dh = &whh_s[row][half * 128];
    #pragma unroll
    for (int q = 0; q < 16; ++q) *(v8bf*)(dh + q * 8) = *(const v8bf*)(gh + q * 8);
    const __bf16* gi = wih + (size_t)grow * DPAD + half * 48;
    __bf16* di = &wih_s[row][half * 48];
    #pragma unroll
    for (int q = 0; q < 6; ++q) *(v8bf*)(di + q * 8) = *(const v8bf*)(gi + q * 8);
  }
  // zero-pad x tile K range [72,96) once; never overwritten by staging
  if (tid < 64)
    for (int d = DIN; d < DPAD; ++d) xs[tid][d] = (__bf16)0.0f;

  // ---- register-resident bias and c state ----------------------------------
  float bias_r[4];
  float creg[8];
  const int col = hc0 + nlo;
  #pragma unroll
  for (int s = 0; s < 4; ++s) bias_r[s] = bias[s * HID + col];
  #pragma unroll
  for (int v = 0; v < 8; ++v)
    creg[v] = c0[(size_t)(b0 + 16 * wv + mhi + v) * HID + col];

  const int xrow = tid >> 1;          // 0..63
  const int xd0  = (tid & 1) * 36;    // 2 threads x 36 floats = 72
  const float* xbase = x + ((size_t)(b0 + xrow) * TSEQ) * DIN + xd0;

  // ---- prologue: load + convert x(0) into packed registers ------------------
  unsigned xreg[18];
  {
    const float* xr = xbase;          // t = 0
    #pragma unroll
    for (int q = 0; q < 18; ++q) {
      BF2 pk;
      pk.b[0] = (__bf16)xr[2 * q];
      pk.b[1] = (__bf16)xr[2 * q + 1];
      xreg[q] = pk.u;
    }
  }

  for (int t = 0; t < TSEQ; ++t) {
    // ---- dump pre-staged x registers into LDS (cheap: 18 dword stores) ------
    {
      unsigned* dst = (unsigned*)&xs[xrow][xd0];
      #pragma unroll
      for (int q = 0; q < 18; ++q) dst[q] = xreg[q];
    }
    __syncthreads();

    // opaque zero: weight-fragment addresses depend on it, so the compiler
    // cannot hoist the (loop-invariant) LDS loads out of the t-loop and spill.
    int opq = 0;
    asm volatile("" : "+v"(opq));
    const __bf16* wsb = &whh_s[0][0] + opq;
    const __bf16* wib = &wih_s[0][0] + opq;

    v8f acc[4];
    #pragma unroll
    for (int s = 0; s < 4; ++s)
      #pragma unroll
      for (int v = 0; v < 8; ++v) acc[s][v] = bias_r[s];

    const __bf16* hread = hbuf + (size_t)(t & 1) * (BAT * HID);
    const __bf16* arow  = hread + (size_t)(b0 + 16 * wv) * HID;

    // ---- B-fragment software pipeline (depth 1) -----------------------------
    v16bf bcur[4], bnxt[4];
    #pragma unroll
    for (int s = 0; s < 4; ++s)
      bcur[s] = load_frag_b(wsb + (size_t)(s * 16) * WHH_LD, WHH_LD, lane);

    // ---- recurrent GEMM: gates += h_t @ W_hh^T (K = 256) --------------------
    #pragma unroll
    for (int kc = 0; kc < 8; ++kc) {
      v16bf a = load_frag_a(arow + kc * 32, HID, lane);
      if (kc < 7) {
        #pragma unroll
        for (int s = 0; s < 4; ++s)
          bnxt[s] = load_frag_b(wsb + (size_t)(s * 16) * WHH_LD + (kc + 1) * 32,
                                WHH_LD, lane);
      } else {
        #pragma unroll
        for (int s = 0; s < 4; ++s)
          bnxt[s] = load_frag_b(wib + (size_t)(s * 16) * WIH_LD, WIH_LD, lane);
      }
      #pragma unroll
      for (int s = 0; s < 4; ++s)
        acc[s] = __builtin_amdgcn_wmma_f32_16x16x32_bf16(
            false, a, false, bcur[s], (short)0, acc[s], false, false);
      #pragma unroll
      for (int s = 0; s < 4; ++s) bcur[s] = bnxt[s];
    }
    // ---- fused input GEMM: gates += x_t @ W_ih^T (K = 96 padded) ------------
    #pragma unroll
    for (int kc = 0; kc < 3; ++kc) {
      v16bf a = load_frag_a(&xs[16 * wv][0] + kc * 32, WIH_LD, lane);
      if (kc < 2) {
        #pragma unroll
        for (int s = 0; s < 4; ++s)
          bnxt[s] = load_frag_b(wib + (size_t)(s * 16) * WIH_LD + (kc + 1) * 32,
                                WIH_LD, lane);
      }
      #pragma unroll
      for (int s = 0; s < 4; ++s)
        acc[s] = __builtin_amdgcn_wmma_f32_16x16x32_bf16(
            false, a, false, bcur[s], (short)0, acc[s], false, false);
      #pragma unroll
      for (int s = 0; s < 4; ++s) bcur[s] = bnxt[s];
    }

    // ---- gates + state update (i,f,g,o register-local) ----------------------
    __bf16* hw = hbuf + (size_t)((t + 1) & 1) * (BAT * HID);
    const bool last = (t == TSEQ - 1);
    #pragma unroll
    for (int v = 0; v < 8; ++v) {
      const int m = b0 + 16 * wv + mhi + v;
      const float ig = fsig (acc[0][v]);
      const float fg = fsig (acc[1][v]);
      const float gg = ftanh(acc[2][v]);
      const float og = fsig (acc[3][v]);
      const float cnew = fg * creg[v] + ig * gg;
      creg[v] = cnew;
      const float hnew = og * ftanh(cnew);
      hw[(size_t)m * HID + col] = (__bf16)hnew;
      if (last) {
        hnf[(size_t)m * HID + col] = hnew;
        cnf[(size_t)m * HID + col] = cnew;
      }
    }

    __threadfence();          // release this thread's h stores (device scope)

    // ---- load + convert x(t+1): overlaps the inter-block barrier ------------
    if (t + 1 < TSEQ) {
      const float* xr = xbase + (size_t)(t + 1) * DIN;
      #pragma unroll
      for (int q = 0; q < 18; ++q) {
        BF2 pk;
        pk.b[0] = (__bf16)xr[2 * q];
        pk.b[1] = (__bf16)xr[2 * q + 1];
        xreg[q] = pk.u;
      }
    }

    // ---- one grid barrier per step ------------------------------------------
    __syncthreads();
    if (tid == 0) {
      atomicAdd(sync_cnt, 1u);
      const unsigned target = (unsigned)(t + 1) * (unsigned)NBLK;
      while (atomicAdd(sync_cnt, 0u) < target) __builtin_amdgcn_s_sleep(1);
    }
    __syncthreads();
    __threadfence();          // acquire other blocks' h stores
  }
}

// ---------------------------------------------------------------------------
// Head: mean = [hn,cn] @ W_mean^T + b_mean ; var likewise (0.13 GFLOP).
// ---------------------------------------------------------------------------
__global__ __launch_bounds__(HID)
void lstm_head_kernel(const float* __restrict__ hn, const float* __restrict__ cn,
                      const float* __restrict__ Wm, const float* __restrict__ bm,
                      const float* __restrict__ Wv, const float* __restrict__ bv,
                      float* __restrict__ out) {
  __shared__ float hid[2 * HID];
  const int b = blockIdx.x, n = threadIdx.x;
  hid[n]       = hn[(size_t)b * HID + n];
  hid[HID + n] = cn[(size_t)b * HID + n];
  __syncthreads();
  float am = bm[n], av = bv[n];
  const float* wmr = Wm + (size_t)n * (2 * HID);
  const float* wvr = Wv + (size_t)n * (2 * HID);
  #pragma unroll 8
  for (int k = 0; k < 2 * HID; ++k) {
    const float h = hid[k];
    am = fmaf(h, wmr[k], am);
    av = fmaf(h, wvr[k], av);
  }
  out[(size_t)b * HID + n]             = am;   // mean [1,B,H]
  out[BAT * HID + (size_t)b * HID + n] = av;   // var  [1,B,H]
}

// ---------------------------------------------------------------------------
// Workspace layout (bytes):
//   0       : sync counter (256)
//   256     : bias fp32 [1024]               (4096)
//   4352    : W_ih bf16 [1024][96]           (196608)
//   200960  : W_hh bf16 [1024][256]          (524288)
//   725248  : h double buffer bf16 [2][B][H] (262144)
//   987392  : hn fp32 [B][H]                 (262144)
//   1249536 : cn fp32 [B][H]                 (262144)   total ~1.45 MB
// ---------------------------------------------------------------------------
extern "C" void kernel_launch(void* const* d_in, const int* in_sizes, int n_in,
                              void* d_out, int out_size, void* d_ws, size_t ws_size,
                              hipStream_t stream) {
  const float* x      = (const float*)d_in[0];
  const float* h0     = (const float*)d_in[1];
  const float* c0     = (const float*)d_in[2];
  const float* W_ih   = (const float*)d_in[3];
  const float* W_hh   = (const float*)d_in[4];
  const float* b_ih   = (const float*)d_in[5];
  const float* b_hh   = (const float*)d_in[6];
  const float* W_mean = (const float*)d_in[7];
  const float* b_mean = (const float*)d_in[8];
  const float* W_var  = (const float*)d_in[9];
  const float* b_var  = (const float*)d_in[10];
  float* out = (float*)d_out;

  char* ws = (char*)d_ws;
  unsigned* sync_cnt = (unsigned*)(ws + 0);
  float*    bias     = (float*)  (ws + 256);
  __bf16*   wih_b    = (__bf16*) (ws + 4352);
  __bf16*   whh_b    = (__bf16*) (ws + 200960);
  __bf16*   hbuf     = (__bf16*) (ws + 725248);
  float*    hnf      = (float*)  (ws + 987392);
  float*    cnf      = (float*)  (ws + 1249536);

  hipMemsetAsync(sync_cnt, 0, 256, stream);  // barrier counter = 0 each call

  lstm_prep_kernel<<<(G4 * HID + 255) / 256, 256, 0, stream>>>(
      W_ih, W_hh, b_ih, b_hh, h0, wih_b, whh_b, bias, hbuf);

  lstm_persistent_kernel<<<NBLK, NTHR, 0, stream>>>(
      x, c0, wih_b, whh_b, bias, hbuf, hnf, cnf, sync_cnt);

  lstm_head_kernel<<<BAT, HID, 0, stream>>>(
      hnf, cnf, W_mean, b_mean, W_var, b_var, out);
}
